// VisionTransformer_20229295964628
// MI455X (gfx1250) — compile-verified, matrix operand order checked by reading, not run on hardware
//
#include <hip/hip_runtime.h>
#include <cstddef>

typedef __bf16        v16bf __attribute__((ext_vector_type(16)));
typedef float         v8f   __attribute__((ext_vector_type(8)));
typedef unsigned int  v8u   __attribute__((ext_vector_type(8)));
typedef int           v4i   __attribute__((ext_vector_type(4)));

#define C_DIM 768
#define HID   3072
#define NHEAD 12
#define DH    64
#define SEQ   800
#define MTOT  3200   // B*S = 4*800

// ---- CDNA5 async LDS copy availability ----
#if defined(__has_builtin)
#if __has_builtin(__builtin_amdgcn_global_load_async_to_lds_b128)
#define USE_ASYNC_LDS 1
#endif
#endif
#ifndef USE_ASYNC_LDS
#define USE_ASYNC_LDS 0
#endif

typedef __attribute__((address_space(1))) v4i* as1_v4i;   // global
typedef __attribute__((address_space(3))) v4i* as3_v4i;   // LDS

__device__ __forceinline__ void wait_async0() {
#if defined(__has_builtin) && __has_builtin(__builtin_amdgcn_s_wait_asynccnt)
    __builtin_amdgcn_s_wait_asynccnt(0);
#else
    asm volatile("s_wait_asynccnt 0" ::: "memory");
#endif
}

// ---------------- helpers ----------------
__device__ __forceinline__ unsigned short f2bf(float f) {
    unsigned int u = __float_as_uint(f);
    unsigned int r = u + 0x7FFFu + ((u >> 16) & 1u);   // RNE
    return (unsigned short)(r >> 16);
}
__device__ __forceinline__ float bf2f(unsigned short u) {
    return __uint_as_float(((unsigned int)u) << 16);
}
// K index (even, of a contiguous pair) for 16-bit A/B fragment, VGPR pair i (0..7)
__device__ __forceinline__ int kmap(int lane, int i) {
    return ((i & 4) << 2) + ((lane & 16) >> 1) + ((i & 3) << 1);
}

// ---------------- im2col for patch embed ----------------
__global__ __launch_bounds__(256)
void im2col_kernel(const float* __restrict__ x, unsigned short* __restrict__ xp) {
    int idx = blockIdx.x * 256 + threadIdx.x;          // MTOT*768 threads
    int row = idx / C_DIM, k = idx % C_DIM;
    int b = row / SEQ, s = row % SEQ;
    int img = (s >= 400) ? 1 : 0;
    int p = s - img * 400;
    int pr = p / 20, pc = p % 20;
    int ch = k >> 8, rem = k & 255, py = rem >> 4, px = rem & 15;
    size_t src = (((size_t)(b * 6 + img * 3 + ch)) * 320 + (size_t)(pr * 16 + py)) * 320
               + (size_t)(pc * 16 + px);
    xp[(size_t)row * C_DIM + k] = f2bf(x[src]);
}

// ---------------- LayerNorm ----------------
template<bool OUTF32>
__global__ __launch_bounds__(256)
void ln_kernel(const float* __restrict__ x, const float* __restrict__ g,
               const float* __restrict__ bta, void* __restrict__ out) {
    __shared__ float red[16];
    int row = blockIdx.x;
    const float* xr = x + (size_t)row * C_DIM;
    int t = threadIdx.x;
    float v0 = xr[t], v1 = xr[t + 256], v2 = xr[t + 512];
    float s = v0 + v1 + v2;
    float ss = v0 * v0 + v1 * v1 + v2 * v2;
    for (int m = 16; m; m >>= 1) { s += __shfl_xor(s, m, 32); ss += __shfl_xor(ss, m, 32); }
    if ((t & 31) == 0) { red[t >> 5] = s; red[8 + (t >> 5)] = ss; }
    __syncthreads();
    if (t == 0) {
        float ts = 0.f, tss = 0.f;
        for (int i = 0; i < 8; ++i) { ts += red[i]; tss += red[8 + i]; }
        red[0] = ts; red[8] = tss;
    }
    __syncthreads();
    float mean = red[0] * (1.0f / 768.0f);
    float var  = red[8] * (1.0f / 768.0f) - mean * mean;
    float rstd = rsqrtf(var + 1e-6f);
    float* of = (float*)out;
    unsigned short* ob = (unsigned short*)out;
    #pragma unroll
    for (int j = 0; j < 3; ++j) {
        int c = t + j * 256;
        float v = (j == 0) ? v0 : (j == 1) ? v1 : v2;
        float y = (v - mean) * rstd * g[c] + bta[c];
        if (OUTF32) of[(size_t)row * C_DIM + c] = y;
        else        ob[(size_t)row * C_DIM + c] = f2bf(y);
    }
}

// ---------------- RoPE (in-place on bf16 qkv) ----------------
__global__ __launch_bounds__(256)
void rope_kernel(unsigned short* __restrict__ qkv) {
    int idx = blockIdx.x * 256 + threadIdx.x;          // MTOT * 768 (q+k pairs)
    int tok = idx / 768, rem = idx % 768;
    int qk = rem / 384, hp = rem % 384;
    int h = hp >> 5, p = hp & 31;
    int s = tok % SEQ;
    unsigned short* ptr = qkv + (size_t)tok * (3 * C_DIM) + qk * C_DIM + h * DH + 2 * p;
    float t0 = bf2f(ptr[0]), t1 = bf2f(ptr[1]);
    const float LOG1E4_32 = 0.28782313662425574f;      // ln(10000)/32
    float f0 = __expf(-LOG1E4_32 * (float)((2 * p) & 31));
    float f1 = __expf(-LOG1E4_32 * (float)((2 * p + 1) & 31));
    float a0 = (float)s * f0, a1 = (float)s * f1;
    float o0 = t0 * __cosf(a0) - t1 * __sinf(a0);
    float o1 = t1 * __cosf(a1) + t0 * __sinf(a1);
    ptr[0] = f2bf(o0); ptr[1] = f2bf(o1);
}

// ---------------- Generic bf16 WMMA GEMM, 64x64 block tile ----------------
// out[m][n] = epi( sum_k A[m][k] * W(k,n) + bias[n] )
// W element (k,n) at W[k*wsK + n*wsN]  (fp32 weights, converted on stage)
#define EPI_STORE_F32  0
#define EPI_STORE_BF16 1
#define EPI_ADD_F32    2
#define EPI_GELU_BF16  3

#define A_STRIDE 40   // halfs; 80B row stride: 16B aligned + bank-conflict-free
#define W_STRIDE 34

template<int EPI>
__global__ __launch_bounds__(128)
void gemm_bf16_wmma(const unsigned short* __restrict__ A, int lda,
                    const float* __restrict__ W, size_t wsK, size_t wsN,
                    const float* __restrict__ bias,
                    void* __restrict__ outp, int ldo, int K) {
    __shared__ unsigned short a_sh[64 * A_STRIDE];
    __shared__ unsigned short w_sh[64 * W_STRIDE];
    const int tid  = threadIdx.x;
    const int lane = tid & 31;
    const int wv   = tid >> 5;
    const int m32  = (wv & 1) * 32;
    const int n32  = (wv >> 1) * 32;
    const int m0 = blockIdx.y * 64;
    const int n0 = blockIdx.x * 64;
    const int l15 = lane & 15;

    v8f c00 = {}, c01 = {}, c10 = {}, c11 = {};

    for (int kk = 0; kk < K; kk += 32) {
        {   // stage A: 64 rows x 32 halfs (two b128 per thread), direct to LDS
            int row = tid >> 1;
            int cg  = (tid & 1) * 16;
            const unsigned short* src = A + (size_t)(m0 + row) * lda + kk + cg;
            unsigned short* dst = &a_sh[row * A_STRIDE + cg];
#if USE_ASYNC_LDS
            __builtin_amdgcn_global_load_async_to_lds_b128((as1_v4i)src,       (as3_v4i)dst,       0, 0);
            __builtin_amdgcn_global_load_async_to_lds_b128((as1_v4i)(src + 8), (as3_v4i)(dst + 8), 0, 0);
#else
            *(uint4*)dst       = *(const uint4*)src;
            *(uint4*)(dst + 8) = *(const uint4*)(src + 8);
#endif
        }
        {   // stage W transposed [n][k], fp32 -> bf16
            int n  = tid >> 1;
            int ks = (tid & 1) * 16;
            size_t base = (size_t)(kk + ks) * wsK + (size_t)(n0 + n) * wsN;
            #pragma unroll 4
            for (int i = 0; i < 16; ++i)
                w_sh[n * W_STRIDE + ks + i] = f2bf(W[base + (size_t)i * wsK]);
            if (kk + 32 < K)   // prefetch next K-slab of weights (RT, speculative)
                __builtin_prefetch(&W[base + 32 * wsK], 0, 0);
        }
#if USE_ASYNC_LDS
        wait_async0();
#endif
        __syncthreads();

        v8u au0, au1, bu0, bu1;
        #pragma unroll
        for (int i = 0; i < 8; ++i) {
            int k = kmap(lane, i);
            au0[i] = *(const unsigned int*)&a_sh[(m32 + l15)      * A_STRIDE + k];
            au1[i] = *(const unsigned int*)&a_sh[(m32 + 16 + l15) * A_STRIDE + k];
            bu0[i] = *(const unsigned int*)&w_sh[(n32 + l15)      * W_STRIDE + k];
            bu1[i] = *(const unsigned int*)&w_sh[(n32 + 16 + l15) * W_STRIDE + k];
        }
        v16bf a0 = __builtin_bit_cast(v16bf, au0);
        v16bf a1 = __builtin_bit_cast(v16bf, au1);
        v16bf b0 = __builtin_bit_cast(v16bf, bu0);
        v16bf b1 = __builtin_bit_cast(v16bf, bu1);
        c00 = __builtin_amdgcn_wmma_f32_16x16x32_bf16(false, a0, false, b0, (short)0, c00, false, false);
        c01 = __builtin_amdgcn_wmma_f32_16x16x32_bf16(false, a0, false, b1, (short)0, c01, false, false);
        c10 = __builtin_amdgcn_wmma_f32_16x16x32_bf16(false, a1, false, b0, (short)0, c10, false, false);
        c11 = __builtin_amdgcn_wmma_f32_16x16x32_bf16(false, a1, false, b1, (short)0, c11, false, false);
        __syncthreads();
    }

    const int mbase = m0 + m32 + ((lane & 16) ? 8 : 0);
    const int nbase = n0 + n32 + l15;
    #pragma unroll
    for (int r = 0; r < 8; ++r) {
        #pragma unroll
        for (int mi = 0; mi < 2; ++mi) {
            #pragma unroll
            for (int gi = 0; gi < 2; ++gi) {
                int m = mbase + mi * 16 + r;
                int n = nbase + gi * 16;
                float v = (mi ? (gi ? c11[r] : c10[r]) : (gi ? c01[r] : c00[r])) + bias[n];
                size_t off = (size_t)m * ldo + n;
                if (EPI == EPI_STORE_F32)       ((float*)outp)[off]  = v;
                else if (EPI == EPI_ADD_F32)    ((float*)outp)[off] += v;
                else if (EPI == EPI_STORE_BF16) ((unsigned short*)outp)[off] = f2bf(v);
                else {
                    float ge = 0.5f * v * (1.0f + erff(v * 0.70710678118f));
                    ((unsigned short*)outp)[off] = f2bf(ge);
                }
            }
        }
    }
}

// ---------------- Flash attention: 1 wave per (b, h, 16-query tile) ----------------
__global__ __launch_bounds__(32)
void attn_kernel(const unsigned short* __restrict__ qkv,
                 unsigned short* __restrict__ obuf) {
    __shared__ unsigned short p_sh[16 * 34];
    const int lane = threadIdx.x;
    int blk = blockIdx.x;
    const int mt = blk % 50; blk /= 50;
    const int h  = blk % NHEAD;
    const int b  = blk / NHEAD;
    const size_t tok0 = (size_t)b * SEQ;
    const int m0   = mt * 16;
    const int lhalf = (lane & 16) ? 8 : 0;
    const int l15   = lane & 15;
    const size_t RS = 3 * C_DIM;   // qkv row stride

    // Q fragments (16 x 64, two K-halves)
    v8u qu0, qu1;
    {
        const unsigned short* qp = qkv + (tok0 + m0 + l15) * RS + h * DH;
        #pragma unroll
        for (int i = 0; i < 8; ++i) {
            int k = kmap(lane, i);
            qu0[i] = *(const unsigned int*)(qp + k);
            qu1[i] = *(const unsigned int*)(qp + 32 + k);
        }
    }
    const v16bf aq0 = __builtin_bit_cast(v16bf, qu0);
    const v16bf aq1 = __builtin_bit_cast(v16bf, qu1);

    float mi[8], li[8];
    v8f o0 = {}, o1 = {}, o2 = {}, o3 = {};
    #pragma unroll
    for (int r = 0; r < 8; ++r) { mi[r] = -3.0e38f; li[r] = 0.0f; }

    for (int kt = 0; kt < SEQ; kt += 32) {
        if (kt + 32 < SEQ)   // prefetch next chunk's K rows
            __builtin_prefetch(qkv + (tok0 + kt + 32 + l15) * RS + C_DIM + h * DH, 0, 0);
        // ---- scores: Q (16x64) x K^T chunk (64x32) -> two 16x16 f32 tiles
        v8f s0 = {}, s1 = {};
        {
            v8u k00, k01, k10, k11;
            const unsigned short* kp0 = qkv + (tok0 + kt + l15) * RS + C_DIM + h * DH;
            const unsigned short* kp1 = kp0 + 16 * RS;
            #pragma unroll
            for (int i = 0; i < 8; ++i) {
                int k = kmap(lane, i);
                k00[i] = *(const unsigned int*)(kp0 + k);
                k01[i] = *(const unsigned int*)(kp0 + 32 + k);
                k10[i] = *(const unsigned int*)(kp1 + k);
                k11[i] = *(const unsigned int*)(kp1 + 32 + k);
            }
            s0 = __builtin_amdgcn_wmma_f32_16x16x32_bf16(false, aq0, false,
                     __builtin_bit_cast(v16bf, k00), (short)0, s0, false, false);
            s0 = __builtin_amdgcn_wmma_f32_16x16x32_bf16(false, aq1, false,
                     __builtin_bit_cast(v16bf, k01), (short)0, s0, false, false);
            s1 = __builtin_amdgcn_wmma_f32_16x16x32_bf16(false, aq0, false,
                     __builtin_bit_cast(v16bf, k10), (short)0, s1, false, false);
            s1 = __builtin_amdgcn_wmma_f32_16x16x32_bf16(false, aq1, false,
                     __builtin_bit_cast(v16bf, k11), (short)0, s1, false, false);
        }
        // ---- online softmax, P -> LDS as bf16 (C-tile layout)
        #pragma unroll
        for (int r = 0; r < 8; ++r) {
            float a  = s0[r] * 0.125f;        // DH^-0.5
            float bb = s1[r] * 0.125f;
            float mx = fmaxf(a, bb);
            mx = fmaxf(mx, __shfl_xor(mx, 1, 32));
            mx = fmaxf(mx, __shfl_xor(mx, 2, 32));
            mx = fmaxf(mx, __shfl_xor(mx, 4, 32));
            mx = fmaxf(mx, __shfl_xor(mx, 8, 32));
            float mnew  = fmaxf(mi[r], mx);
            float alpha = __expf(mi[r] - mnew);
            float p0 = __expf(a - mnew);
            float p1 = __expf(bb - mnew);
            float rs = p0 + p1;
            rs += __shfl_xor(rs, 1, 32);
            rs += __shfl_xor(rs, 2, 32);
            rs += __shfl_xor(rs, 4, 32);
            rs += __shfl_xor(rs, 8, 32);
            li[r] = li[r] * alpha + rs;
            mi[r] = mnew;
            o0[r] *= alpha; o1[r] *= alpha; o2[r] *= alpha; o3[r] *= alpha;
            int mrow = r + lhalf;
            p_sh[mrow * 34 + l15]      = f2bf(p0);
            p_sh[mrow * 34 + 16 + l15] = f2bf(p1);
        }
        asm volatile("s_wait_dscnt 0" ::: "memory");   // wave-local LDS RAW
        // ---- re-layout P as A-fragment (16 rows x 32 keys)
        v8u pu;
        #pragma unroll
        for (int i = 0; i < 8; ++i)
            pu[i] = *(const unsigned int*)&p_sh[l15 * 34 + kmap(lane, i)];
        const v16bf pf = __builtin_bit_cast(v16bf, pu);
        asm volatile("" ::: "memory");
        // ---- V fragments: B[k][n] = V[kt+k][nt*16+n]
        const unsigned short* vp = qkv + (tok0 + kt) * RS + 2 * C_DIM + h * DH + l15;
        v8u v0u, v1u, v2u, v3u;
        #pragma unroll
        for (int i = 0; i < 8; ++i) {
            int k = kmap(lane, i);
            const unsigned short* pa = vp + (size_t)k * RS;
            const unsigned short* pb = pa + RS;
            v0u[i] = (unsigned int)pa[0]  | ((unsigned int)pb[0]  << 16);
            v1u[i] = (unsigned int)pa[16] | ((unsigned int)pb[16] << 16);
            v2u[i] = (unsigned int)pa[32] | ((unsigned int)pb[32] << 16);
            v3u[i] = (unsigned int)pa[48] | ((unsigned int)pb[48] << 16);
        }
        o0 = __builtin_amdgcn_wmma_f32_16x16x32_bf16(false, pf, false,
                 __builtin_bit_cast(v16bf, v0u), (short)0, o0, false, false);
        o1 = __builtin_amdgcn_wmma_f32_16x16x32_bf16(false, pf, false,
                 __builtin_bit_cast(v16bf, v1u), (short)0, o1, false, false);
        o2 = __builtin_amdgcn_wmma_f32_16x16x32_bf16(false, pf, false,
                 __builtin_bit_cast(v16bf, v2u), (short)0, o2, false, false);
        o3 = __builtin_amdgcn_wmma_f32_16x16x32_bf16(false, pf, false,
                 __builtin_bit_cast(v16bf, v3u), (short)0, o3, false, false);
    }
    // ---- write O (normalize by row sums), bf16 for proj GEMM
    #pragma unroll
    for (int r = 0; r < 8; ++r) {
        float inv = 1.0f / li[r];
        size_t row = tok0 + m0 + r + lhalf;
        unsigned short* op = obuf + row * C_DIM + h * DH + l15;
        op[0]  = f2bf(o0[r] * inv);
        op[16] = f2bf(o1[r] * inv);
        op[32] = f2bf(o2[r] * inv);
        op[48] = f2bf(o3[r] * inv);
    }
}

// ---------------- launcher ----------------
extern "C" void kernel_launch(void* const* d_in, const int* in_sizes, int n_in,
                              void* d_out, int out_size, void* d_ws, size_t ws_size,
                              hipStream_t stream) {
    const float* x      = (const float*)d_in[0];
    const float* conv_w = (const float*)d_in[1];
    const float* conv_b = (const float*)d_in[2];
    const float* ln1_g  = (const float*)d_in[3];
    const float* ln1_b  = (const float*)d_in[4];
    const float* qkv_w  = (const float*)d_in[5];
    const float* qkv_b  = (const float*)d_in[6];
    const float* proj_w = (const float*)d_in[7];
    const float* proj_b = (const float*)d_in[8];
    const float* ln2_g  = (const float*)d_in[9];
    const float* ln2_b  = (const float*)d_in[10];
    const float* fc1_w  = (const float*)d_in[11];
    const float* fc1_b  = (const float*)d_in[12];
    const float* fc2_w  = (const float*)d_in[13];
    const float* fc2_b  = (const float*)d_in[14];
    const float* lnf_g  = (const float*)d_in[15];
    const float* lnf_b  = (const float*)d_in[16];

    char* ws = (char*)d_ws;
    float*          hbuf = (float*)ws;                          //  9,830,400 B
    unsigned short* ybuf = (unsigned short*)(ws +  9830400);    //  4,915,200 B
    unsigned short* qkvb = (unsigned short*)(ws + 14745600);    // 14,745,600 B
    unsigned short* obuf = (unsigned short*)(ws + 29491200);    //  4,915,200 B
    unsigned short* hidb = (unsigned short*)(ws + 34406400);    // 19,660,800 B

    const dim3 g768(12, 50), g2304(36, 50), g3072(48, 50);

    // patch embed: im2col + 768x768 GEMM (conv_w is out-major: wsK=1, wsN=768)
    im2col_kernel<<<9600, 256, 0, stream>>>(x, ybuf);
    gemm_bf16_wmma<EPI_STORE_F32><<<g768, 128, 0, stream>>>(
        ybuf, C_DIM, conv_w, 1, C_DIM, conv_b, hbuf, C_DIM, C_DIM);

    for (int d = 0; d < 12; ++d) {
        ln_kernel<false><<<MTOT, 256, 0, stream>>>(hbuf, ln1_g + d * C_DIM, ln1_b + d * C_DIM, ybuf);
        gemm_bf16_wmma<EPI_STORE_BF16><<<g2304, 128, 0, stream>>>(
            ybuf, C_DIM, qkv_w + (size_t)d * C_DIM * 3 * C_DIM, 3 * C_DIM, 1,
            qkv_b + (size_t)d * 3 * C_DIM, qkvb, 3 * C_DIM, C_DIM);
        rope_kernel<<<9600, 256, 0, stream>>>(qkvb);
        attn_kernel<<<4 * NHEAD * 50, 32, 0, stream>>>(qkvb, obuf);
        gemm_bf16_wmma<EPI_ADD_F32><<<g768, 128, 0, stream>>>(
            obuf, C_DIM, proj_w + (size_t)d * C_DIM * C_DIM, C_DIM, 1,
            proj_b + (size_t)d * C_DIM, hbuf, C_DIM, C_DIM);
        ln_kernel<false><<<MTOT, 256, 0, stream>>>(hbuf, ln2_g + d * C_DIM, ln2_b + d * C_DIM, ybuf);
        gemm_bf16_wmma<EPI_GELU_BF16><<<g3072, 128, 0, stream>>>(
            ybuf, C_DIM, fc1_w + (size_t)d * C_DIM * HID, HID, 1,
            fc1_b + (size_t)d * HID, hidb, HID, C_DIM);
        gemm_bf16_wmma<EPI_ADD_F32><<<g768, 128, 0, stream>>>(
            hidb, HID, fc2_w + (size_t)d * HID * C_DIM, C_DIM, 1,
            fc2_b + (size_t)d * C_DIM, hbuf, C_DIM, HID);
    }
    ln_kernel<true><<<MTOT, 256, 0, stream>>>(hbuf, lnf_g, lnf_b, d_out);
}
